// myNNConv_2327872274896
// MI455X (gfx1250) — matrix operand order repaired and checked by reading, not compiled, not run on hardware
//
#include <hip/hip_runtime.h>

#define IN_C   128
#define TWO_C  256
#define OUT_C  128
#define TILE_E 64
#define BLK    256
#define BN_EPS 1e-5f

typedef __bf16 bf16x16 __attribute__((ext_vector_type(16)));
typedef float  f32x8   __attribute__((ext_vector_type(8)));

union FragAB { uint4 u[2]; bf16x16 v; };

// float -> bf16 bits, round-to-nearest-even
__device__ __forceinline__ unsigned f2bf(float f) {
  unsigned u = __float_as_uint(f);
  u += 0x7FFFu + ((u >> 16) & 1u);
  return (u >> 16) & 0xFFFFu;
}

// Order-preserving float <-> uint map for atomic max.
// enc==0 is an "empty" sentinel (cannot be produced by any finite float).
__device__ __forceinline__ unsigned encMax(float f) {
  unsigned u = __float_as_uint(f);
  return (u & 0x80000000u) ? ~u : (u | 0x80000000u);
}
__device__ __forceinline__ float decMax(unsigned e) {
  unsigned u = (e & 0x80000000u) ? (e & 0x7FFFFFFFu) : ~e;
  return __uint_as_float(u);
}

__global__ __launch_bounds__(BLK) void init_ws_kernel(unsigned* p, long n) {
  long i = (long)blockIdx.x * BLK + threadIdx.x;
  long stride = (long)gridDim.x * BLK;
  for (; i < n; i += stride) p[i] = 0u;
}

// ---------------------------------------------------------------------------
// Pass 1: per-edge MLP via WMMA bf16 (f32 accumulate) + scatter-max (encoded)
// ---------------------------------------------------------------------------
__global__ __launch_bounds__(BLK) void edgeconv_kernel(
    const float* __restrict__ x, const int* __restrict__ ei,
    const float* __restrict__ W1, const float* __restrict__ b1,
    const float* __restrict__ W2, const float* __restrict__ b2,
    unsigned* __restrict__ agg_enc, int nEdges)
{
  // LDS: W1 frags 64KB + W2 frags 32KB + A-stage 33KB + H-stage 17KB + idx
  __shared__ alignas(16) unsigned       sW1[16384];        // 8 ksteps * 8 ntiles * 32 lanes * 8 dwords
  __shared__ alignas(16) unsigned       sW2[8192];         // 4 ksteps * ...
  __shared__ alignas(16) unsigned short sM[TILE_E * 264];  // 64 rows x 256 bf16 (+8 pad)
  __shared__ alignas(16) unsigned short sH[TILE_E * 136];  // 64 rows x 128 bf16 (+8 pad)
  __shared__ int sSrc[TILE_E];
  __shared__ int sDst[TILE_E];

  const int tid  = threadIdx.x;
  const int wave = tid >> 5;
  const int lane = tid & 31;
  const int nlo  = lane & 15;
  const int ncol = wave * 16 + nlo;     // this wave's output column for C/D frags
  const int hiK  = (lane & 16) ? 1 : 0; // lane in upper half of wave

  // --- Swizzle weights (f32 global -> bf16 B-fragment order in LDS), once ---
  // B frag layout (32x16 bf16): VGPR v<4: K=8h+2v,+1; VGPR v>=4: K=16+8h+2(v-4),+1 (h=lane>=16)
  for (int f = tid; f < 16384; f += BLK) {
    int v  = f & 7;
    int ln = (f >> 3) & 31;
    int nt = (f >> 8) & 7;
    int ks = f >> 11;
    int k  = ks * 32 + 2 * (v & 3) + ((v & 4) ? 16 : 0) + ((ln & 16) ? 8 : 0);
    int n  = nt * 16 + (ln & 15);
    sW1[f] = f2bf(W1[k * OUT_C + n]) | (f2bf(W1[(k + 1) * OUT_C + n]) << 16);
  }
  for (int f = tid; f < 8192; f += BLK) {
    int v  = f & 7;
    int ln = (f >> 3) & 31;
    int nt = (f >> 8) & 7;
    int ks = f >> 11;
    int k  = ks * 32 + 2 * (v & 3) + ((v & 4) ? 16 : 0) + ((ln & 16) ? 8 : 0);
    int n  = nt * 16 + (ln & 15);
    sW2[f] = f2bf(W2[k * OUT_C + n]) | (f2bf(W2[(k + 1) * OUT_C + n]) << 16);
  }
  const float b1v = b1[ncol];
  const float b2v = b2[ncol];
  __syncthreads();

  const int nTiles = (nEdges + TILE_E - 1) / TILE_E;
  for (int t = blockIdx.x; t < nTiles; t += gridDim.x) {
    const int e0 = t * TILE_E;

    // --- load edge indices for this tile ---
    if (tid < TILE_E) {
      int e  = e0 + tid;
      int ec = (e < nEdges) ? e : 0;
      sSrc[tid] = ei[ec];
      sDst[tid] = ei[nEdges + ec];
    }
    __syncthreads();

    // --- gather m = [x_i | x_j - x_i] into LDS as bf16 ---
    for (int q = tid; q < TILE_E * 32; q += BLK) {
      int e = q >> 5;
      int c = (q & 31) << 2;
      const float4 xi = *(const float4*)(x + (long)sDst[e] * IN_C + c);
      const float4 xj = *(const float4*)(x + (long)sSrc[e] * IN_C + c);
      uint2 pi, pd;
      pi.x = f2bf(xi.x) | (f2bf(xi.y) << 16);
      pi.y = f2bf(xi.z) | (f2bf(xi.w) << 16);
      pd.x = f2bf(xj.x - xi.x) | (f2bf(xj.y - xi.y) << 16);
      pd.y = f2bf(xj.z - xi.z) | (f2bf(xj.w - xi.w) << 16);
      *(uint2*)&sM[e * 264 + c]        = pi;
      *(uint2*)&sM[e * 264 + IN_C + c] = pd;
    }
    __syncthreads();

    // --- Stage 1: h1[64x128] = relu(m[64x256] @ W1 + b1) ---
    // A frag (16x32 bf16): lane<16: row=nlo, K 0-7 & 16-23; lane>=16: row=nlo, K 8-15 & 24-31
    f32x8 acc[4];
#pragma unroll
    for (int mt = 0; mt < 4; ++mt) acc[mt] = (f32x8)(0.0f);
#pragma unroll
    for (int ks = 0; ks < 8; ++ks) {
      FragAB B;
      const uint4* bp = (const uint4*)&sW1[((ks * 8 + wave) * 32 + lane) * 8];
      B.u[0] = bp[0];
      B.u[1] = bp[1];
      const int ko = ks * 32 + hiK * 8;  // element offset within row
#pragma unroll
      for (int mt = 0; mt < 4; ++mt) {
        const unsigned short* rp = &sM[(mt * 16 + nlo) * 264 + ko];
        FragAB A;
        A.u[0] = *(const uint4*)(rp);
        A.u[1] = *(const uint4*)(rp + 16);
        acc[mt] = __builtin_amdgcn_wmma_f32_16x16x32_bf16(
            false, A.v, false, B.v, (short)0, acc[mt], false, false);
      }
    }
    // bias + relu -> sH (bf16). C/D frag: VGPR r -> M=r (+8 for upper lanes), N=nlo
#pragma unroll
    for (int mt = 0; mt < 4; ++mt) {
#pragma unroll
      for (int r = 0; r < 8; ++r) {
        int row = mt * 16 + r + hiK * 8;
        float h = acc[mt][r] + b1v;
        h = h > 0.0f ? h : 0.0f;
        sH[row * 136 + ncol] = (unsigned short)f2bf(h);
      }
    }
    __syncthreads();

    // --- Stage 2: out[64x128] = h1 @ W2 + b2 ---
    f32x8 acc2[4];
#pragma unroll
    for (int mt = 0; mt < 4; ++mt) acc2[mt] = (f32x8)(0.0f);
#pragma unroll
    for (int ks = 0; ks < 4; ++ks) {
      FragAB B;
      const uint4* bp = (const uint4*)&sW2[((ks * 8 + wave) * 32 + lane) * 8];
      B.u[0] = bp[0];
      B.u[1] = bp[1];
      const int ko = ks * 32 + hiK * 8;
#pragma unroll
      for (int mt = 0; mt < 4; ++mt) {
        const unsigned short* rp = &sH[(mt * 16 + nlo) * 136 + ko];
        FragAB A;
        A.u[0] = *(const uint4*)(rp);
        A.u[1] = *(const uint4*)(rp + 16);
        acc2[mt] = __builtin_amdgcn_wmma_f32_16x16x32_bf16(
            false, A.v, false, B.v, (short)0, acc2[mt], false, false);
      }
    }

    // --- scatter-max (order-preserving encoding, u32 atomicMax) ---
#pragma unroll
    for (int mt = 0; mt < 4; ++mt) {
#pragma unroll
      for (int r = 0; r < 8; ++r) {
        int row = mt * 16 + r + hiK * 8;
        int e   = e0 + row;
        if (e < nEdges) {
          float v = acc2[mt][r] + b2v;
          atomicMax(&agg_enc[(unsigned long long)sDst[row] * OUT_C + ncol], encMax(v));
        }
      }
    }
    __syncthreads();
  }
}

// ---------------------------------------------------------------------------
// Pass 2: decode encoded max in place (empty -> 0), per-channel sum & sumsq
// ---------------------------------------------------------------------------
__global__ __launch_bounds__(BLK) void stats_kernel(unsigned* __restrict__ agg,
                                                    float* __restrict__ stats,
                                                    int nNodes)
{
  const int col  = threadIdx.x & 127;
  const int half = threadIdx.x >> 7;
  float s = 0.0f, sq = 0.0f;
  for (int r = blockIdx.x * 2 + half; r < nNodes; r += gridDim.x * 2) {
    long idx = (long)r * OUT_C + col;
    unsigned e = agg[idx];
    float v = (e == 0u) ? 0.0f : decMax(e);
    if (!isfinite(v)) v = 0.0f;
    ((float*)agg)[idx] = v;
    s  += v;
    sq += v * v;
  }
  __shared__ float ss[BLK], sb[BLK];
  ss[threadIdx.x] = s;
  sb[threadIdx.x] = sq;
  __syncthreads();
  if (threadIdx.x < 128) {
    atomicAdd(&stats[col],       ss[threadIdx.x] + ss[threadIdx.x + 128]);
    atomicAdd(&stats[128 + col], sb[threadIdx.x] + sb[threadIdx.x + 128]);
  }
}

// ---------------------------------------------------------------------------
// Pass 3: BatchNorm (training-mode batch stats, biased variance)
// ---------------------------------------------------------------------------
__global__ __launch_bounds__(BLK) void norm_kernel(const float* __restrict__ agg,
                                                   const float* __restrict__ stats,
                                                   const float* __restrict__ gamma,
                                                   const float* __restrict__ beta,
                                                   float* __restrict__ out,
                                                   int nNodes)
{
  const float invN = 1.0f / (float)nNodes;
  long total  = (long)nNodes * OUT_C;
  long i      = (long)blockIdx.x * BLK + threadIdx.x;
  long stride = (long)gridDim.x * BLK;
  for (; i < total; i += stride) {
    int col = (int)(i & (OUT_C - 1));
    float mean = stats[col] * invN;
    float var  = stats[128 + col] * invN - mean * mean;
    float rs   = rsqrtf(var + BN_EPS);
    out[i] = (agg[i] - mean) * rs * gamma[col] + beta[col];
  }
}

extern "C" void kernel_launch(void* const* d_in, const int* in_sizes, int n_in,
                              void* d_out, int out_size, void* d_ws, size_t ws_size,
                              hipStream_t stream) {
  (void)n_in; (void)out_size; (void)ws_size;
  const float* x     = (const float*)d_in[0];
  const int*   ei    = (const int*)d_in[1];
  const float* W1    = (const float*)d_in[2];
  const float* b1    = (const float*)d_in[3];
  const float* W2    = (const float*)d_in[4];
  const float* b2    = (const float*)d_in[5];
  const float* gamma = (const float*)d_in[6];
  const float* beta  = (const float*)d_in[7];

  const int nNodes = in_sizes[0] / IN_C;
  const int nEdges = in_sizes[1] / 2;

  unsigned* agg   = (unsigned*)d_ws;                                   // nNodes*128 u32
  float*    stats = (float*)((char*)d_ws + (size_t)nNodes * OUT_C * 4); // 256 f32

  long initN = (long)nNodes * OUT_C + 256;
  init_ws_kernel<<<1024, BLK, 0, stream>>>((unsigned*)d_ws, initN);

  int nTiles = (nEdges + TILE_E - 1) / TILE_E;
  int grid   = nTiles < 2500 ? nTiles : 2500;
  edgeconv_kernel<<<grid, BLK, 0, stream>>>(x, ei, W1, b1, W2, b2, agg, nEdges);

  stats_kernel<<<512, BLK, 0, stream>>>(agg, stats, nNodes);
  norm_kernel<<<2048, BLK, 0, stream>>>((const float*)agg, stats, gamma, beta,
                                        (float*)d_out, nNodes);
}
